// LSTMText_62869731279588
// MI455X (gfx1250) — matrix operand order repaired and checked by reading, not compiled, not run on hardware
//
#include <hip/hip_runtime.h>
#include <hip/hip_bf16.h>

// ---------------------------------------------------------------------------
// 2-layer LSTM (T=512,B=2048,IN=27,H=40) + linear head (40->27) for gfx1250.
// Persistent batch-tile kernels: each wave32 owns 16 batch rows, keeps h/c in
// LDS across all 512 steps, computes the fused gate GEMM
//     g = [x_t | h] @ [W_ih | W_hh]^T
// with v_wmma_f32_16x16x32_f16 (f16 operands, K zero-padded, f32 accum).
// Input streams are double-buffered one timestep ahead with async global->LDS
// copies (ASYNCcnt) when the toolchain exposes them. Layer 1 fuses the output
// projection. hs1 flows between layers as f16 in workspace.
// ---------------------------------------------------------------------------

#define TT  512
#define BB  2048
#define IN_ 27
#define HH  40
#define OUT_ 27
#define GG  160           // 4*H

typedef __attribute__((ext_vector_type(16))) _Float16 v16h;
typedef __attribute__((ext_vector_type(8)))  _Float16 v8h;
typedef __attribute__((ext_vector_type(8)))  float    v8f;
typedef __attribute__((ext_vector_type(4)))  int      v4i;

#ifndef __has_builtin
#define __has_builtin(x) 0
#endif

#if defined(__HIP_DEVICE_COMPILE__) && \
    __has_builtin(__builtin_amdgcn_global_load_async_to_lds_b32) && \
    __has_builtin(__builtin_amdgcn_global_load_async_to_lds_b128)
#define ASYNC_STAGE 1
#else
#define ASYNC_STAGE 0
#endif

#if ASYNC_STAGE
// Signatures (from clang diagnostics on this toolchain):
//   b32 : (__device__ int*  src, __shared__ int*  dst, Imm offset, Imm cpol)
//   b128: (__device__ int4* src, __shared__ int4* dst, Imm offset, Imm cpol)
typedef __attribute__((address_space(1))) int  g_i32;
typedef __attribute__((address_space(3))) int  l_i32;
typedef __attribute__((address_space(1))) v4i  g_v4i;
typedef __attribute__((address_space(3))) v4i  l_v4i;
static __device__ __forceinline__ void async_cp_b32(const void* g, void* l) {
  __builtin_amdgcn_global_load_async_to_lds_b32((g_i32*)g, (l_i32*)l, 0, 0);
}
static __device__ __forceinline__ void async_cp_b128(const void* g, void* l) {
  __builtin_amdgcn_global_load_async_to_lds_b128((g_v4i*)g, (l_v4i*)l, 0, 0);
}
#endif

static __device__ __forceinline__ void wait_stage() {
#if ASYNC_STAGE
#if __has_builtin(__builtin_amdgcn_s_wait_asynccnt)
  __builtin_amdgcn_s_wait_asynccnt(0);
#else
  asm volatile("s_wait_asynccnt 0x0" ::: "memory");
#endif
#else
  __syncthreads();
#endif
}

static __device__ __forceinline__ v16h make16(v8h lo, v8h hi) {
  return __builtin_shufflevector(lo, hi, 0,1,2,3,4,5,6,7,8,9,10,11,12,13,14,15);
}

// A-fragment (16x32 f16, wave32): lane holds row (lane&15), K runs
// {koff..koff+7} and {koff+16..koff+23}, koff = (lane>=16)?8:0.
static __device__ __forceinline__ v16h fragA(const _Float16* rowp, int kbase, int koff) {
  const _Float16* p = rowp + kbase + koff;
  return make16(*(const v8h*)p, *(const v8h*)(p + 16));
}

// Same A-fragment, but source row is f32 in LDS; convert while assembling.
static __device__ __forceinline__ v16h fragA_f32(const float* rowp, int koff) {
  v16h r;
#pragma unroll
  for (int i = 0; i < 8; ++i) {
    r[i]     = (_Float16)rowp[koff + i];
    r[i + 8] = (_Float16)rowp[koff + 16 + i];
  }
  return r;
}

// B-fragment (32x16 f16): lane holds column (lane&15), K runs
// {koff..koff+7} and {koff+16..koff+23}, koff = (lane>=16)?8:0. p = chunk + koff.
static __device__ __forceinline__ v16h fragB(const _Float16* p) {
  return make16(*(const v8h*)p, *(const v8h*)(p + 16));
}

static __device__ __forceinline__ v8f wmma_f16(v16h a, v16h b, v8f c) {
  return __builtin_amdgcn_wmma_f32_16x16x32_f16(false, a, false, b, (short)0, c, false, false);
}

static __device__ __forceinline__ float sigf(float v) {
  return 1.0f / (1.0f + __expf(-v));
}

// ---------------------------------------------------------------------------
// Weight packing: f16, zero-padded-K, row-major [N][Kpad] so a B-fragment is
// two 16B LDS loads.
//   layer0: Kpad=96  : k<27 -> W_ih0[n][k];      32<=k<72  -> W_hh0[n][k-32]
//   layer1: Kpad=128 : k<40 -> W_ih1[n][k];      64<=k<104 -> W_hh1[n][k-64]
//   head  : [32][64] : n<27 && k<40 -> W_out[n][k]
// ---------------------------------------------------------------------------
__global__ void prep_kernel(const float* __restrict__ Wih0, const float* __restrict__ Whh0,
                            const float* __restrict__ bih0, const float* __restrict__ bhh0,
                            const float* __restrict__ Wih1, const float* __restrict__ Whh1,
                            const float* __restrict__ bih1, const float* __restrict__ bhh1,
                            const float* __restrict__ Wout, const float* __restrict__ bout,
                            _Float16* __restrict__ wc0, _Float16* __restrict__ wc1,
                            _Float16* __restrict__ wo,  float* __restrict__ b0,
                            float* __restrict__ b1,     float* __restrict__ bo) {
  int tid = blockIdx.x * blockDim.x + threadIdx.x;
  int np  = gridDim.x * blockDim.x;
  for (int i = tid; i < GG * 96; i += np) {
    int n = i / 96, k = i - n * 96;
    float v = 0.f;
    if (k < IN_)                 v = Wih0[n * IN_ + k];
    else if (k >= 32 && k < 72)  v = Whh0[n * HH + (k - 32)];
    wc0[i] = (_Float16)v;
  }
  for (int i = tid; i < GG * 128; i += np) {
    int n = i / 128, k = i - n * 128;
    float v = 0.f;
    if (k < HH)                  v = Wih1[n * HH + k];
    else if (k >= 64 && k < 104) v = Whh1[n * HH + (k - 64)];
    wc1[i] = (_Float16)v;
  }
  for (int i = tid; i < 32 * 64; i += np) {
    int n = i / 64, k = i - n * 64;
    float v = (n < OUT_ && k < HH) ? Wout[n * HH + k] : 0.f;
    wo[i] = (_Float16)v;
  }
  for (int i = tid; i < GG; i += np) {
    b0[i] = bih0[i] + bhh0[i];
    b1[i] = bih1[i] + bhh1[i];
  }
  for (int i = tid; i < 32; i += np) bo[i] = (i < OUT_) ? bout[i] : 0.f;
}

// ---------------------------------------------------------------------------
// Input staging into double-buffered LDS, one timestep ahead.
//   layer0: raw f32 rows (27 wide) into [16][32] f32 (pad cols stay zero)
//   layer1: f16 rows (40 wide, 80B = 5 x 16B chunks) into [16][64] f16
// ---------------------------------------------------------------------------
template <int LAYER>
static __device__ __forceinline__ void stage_input(int t, int buf, int lane, int m0,
                                                   const float* __restrict__ x,
                                                   const _Float16* __restrict__ hin,
                                                   float* lXf, _Float16* lXh) {
  if (LAYER == 0) {
    for (int c = lane; c < 16 * IN_; c += 32) {
      int r = c / IN_, k = c - r * IN_;
      const float* g = x + (size_t)((t * BB + m0 + r) * IN_ + k);
      float* l = &lXf[buf * (16 * 32) + r * 32 + k];
#if ASYNC_STAGE
      async_cp_b32(g, l);
#else
      *l = *g;
#endif
    }
  } else {
    for (int c = lane; c < 16 * 5; c += 32) {          // 5 x 16B chunks per row
      int r = c / 5, p = c - r * 5;
      const _Float16* g = hin + (size_t)(t * BB + m0 + r) * HH + p * 8;
      _Float16* l = &lXh[buf * (16 * 64) + r * 64 + p * 8];
#if ASYNC_STAGE
      async_cp_b128(g, l);
#else
      *(v8h*)l = *(const v8h*)g;
#endif
    }
  }
}

// ---------------------------------------------------------------------------
// Persistent recurrent kernel. One wave32 per 16 batch rows; loops t=0..511.
// ---------------------------------------------------------------------------
template <int LAYER>
__global__ __launch_bounds__(32)
void lstm_layer_kernel(const float* __restrict__ x,        // LAYER==0 input (T,B,IN) f32
                       const _Float16* __restrict__ hin,   // LAYER==1 input hs1 (T,B,H) f16
                       const _Float16* __restrict__ wcomb, // packed [G][KS] f16
                       const float* __restrict__ bias,     // [G] = b_ih + b_hh
                       const _Float16* __restrict__ wout,  // [32][64] f16 (LAYER==1)
                       const float* __restrict__ boutp,    // [32] (LAYER==1)
                       _Float16* __restrict__ hs_out,      // LAYER==0: hs1 f16 out
                       float* __restrict__ y,              // LAYER==1: (T,B,OUT) f32
                       float* __restrict__ outH,           // final h for this layer (B,H)
                       float* __restrict__ outC) {         // final c for this layer (B,H)
  constexpr int KS = (LAYER == 0) ? 96 : 128;   // padded K of fused gate GEMM
  constexpr int HB = (LAYER == 0) ? 32 : 64;    // K offset where W_hh chunks start

  __shared__ alignas(16) _Float16 lW[GG * KS];                      // packed weights
  __shared__ alignas(16) float    lXf[(LAYER == 0) ? 2 * 16 * 32 : 8]; // x double-buffer (f32)
  __shared__ alignas(16) _Float16 lXh[(LAYER == 1) ? 2 * 16 * 64 : 8]; // hs1 double-buffer (f16)
  __shared__ alignas(16) _Float16 lH[16 * 64];                      // h state (f16, K-pad 64)
  __shared__ alignas(16) _Float16 lWo[(LAYER == 1) ? 32 * 64 : 8];
  __shared__ float lG[16 * GG];                                     // raw gate accumulators
  __shared__ float lC[16 * HH];                                     // c state (f32)
  __shared__ float lB[GG];
  __shared__ float lbo[(LAYER == 1) ? 32 : 1];

  const int lane = threadIdx.x;
  const int m0   = blockIdx.x * 16;

  // Stage weights/biases into LDS (once).
  {
    v8h* d = (v8h*)lW;
    const v8h* s = (const v8h*)wcomb;
    for (int i = lane; i < GG * KS / 8; i += 32) d[i] = s[i];
    for (int i = lane; i < GG; i += 32) lB[i] = bias[i];
    if (LAYER == 1) {
      v8h* dw = (v8h*)lWo;
      const v8h* sw = (const v8h*)wout;
      for (int i = lane; i < 32 * 64 / 8; i += 32) dw[i] = sw[i];
      for (int i = lane; i < 32; i += 32) lbo[i] = boutp[i];
    }
  }
  // Zero h/c state and input staging buffers (K-pads stay zero forever).
  for (int i = lane; i < 16 * 64; i += 32) lH[i] = (_Float16)0.f;
  for (int i = lane; i < 16 * HH; i += 32) lC[i] = 0.f;
  if (LAYER == 0) for (int i = lane; i < 2 * 16 * 32; i += 32) lXf[i] = 0.f;
  if (LAYER == 1) for (int i = lane; i < 2 * 16 * 64; i += 32) lXh[i] = (_Float16)0.f;
  __syncthreads();

  // Kick off staging of t=0 into buffer 0.
  stage_input<LAYER>(0, 0, lane, m0, x, hin, lXf, lXh);

  const int row  = lane & 15;
  const int koff = (lane >> 4) * 8;    // A-fragment half-row offset
  const int kl   = (lane >> 4) * 8;    // B-fragment K offset within a chunk
  const int nloc = lane & 15;          // B-fragment column within tile
  const int mb   = (lane >> 4) * 8;    // D-fragment row base

  // Loop-invariant output-projection B fragments held in registers.
  v16h Bo00, Bo01, Bo10, Bo11;
  if (LAYER == 1) {
    const _Float16* w0 = lWo + nloc * 64;
    const _Float16* w1 = lWo + (16 + nloc) * 64;
    Bo00 = fragB(w0 + kl);
    Bo01 = fragB(w0 + 32 + kl);
    Bo10 = fragB(w1 + kl);
    Bo11 = fragB(w1 + 32 + kl);
  }

  for (int t = 0; t < TT; ++t) {
    wait_stage();                       // data for step t staged & visible
    const int cur = t & 1;
    if (t + 1 < TT)                     // fire-and-forget stage of t+1
      stage_input<LAYER>(t + 1, cur ^ 1, lane, m0, x, hin, lXf, lXh);

    // ---- A fragments --------------------------------------------------------
    v16h Ax0, Ax1;
    if (LAYER == 0) {
      Ax0 = fragA_f32(&lXf[cur * (16 * 32) + row * 32], koff);
      Ax1 = Ax0;  // unused
    } else {
      const _Float16* xr = &lXh[cur * (16 * 64) + row * 64];
      Ax0 = fragA(xr, 0, koff);
      Ax1 = fragA(xr, 32, koff);
    }
    v16h Ah0 = fragA(lH + row * 64, 0, koff);
    v16h Ah1 = fragA(lH + row * 64, 32, koff);

    // ---- fused gate GEMM: 10 N-tiles of 16 ---------------------------------
#pragma unroll
    for (int j = 0; j < GG / 16; ++j) {
      const _Float16* wb = lW + (16 * j + nloc) * KS;
      v8f acc = {};
      acc = wmma_f16(Ax0, fragB(wb + kl), acc);
      if (LAYER == 1) acc = wmma_f16(Ax1, fragB(wb + 32 + kl), acc);
      acc = wmma_f16(Ah0, fragB(wb + HB + kl), acc);
      acc = wmma_f16(Ah1, fragB(wb + HB + 32 + kl), acc);
#pragma unroll
      for (int r = 0; r < 8; ++r) lG[(mb + r) * GG + 16 * j + nloc] = acc[r];
    }
    __syncthreads();

    // ---- elementwise LSTM cell update (i,f,g,o gate order) -----------------
    for (int e = lane; e < 16 * HH; e += 32) {
      int m = e / HH, n = e - m * HH;
      float gi = lG[m * GG + n]          + lB[n];
      float gf = lG[m * GG + HH + n]     + lB[HH + n];
      float gg = lG[m * GG + 2 * HH + n] + lB[2 * HH + n];
      float go = lG[m * GG + 3 * HH + n] + lB[3 * HH + n];
      float c  = sigf(gf) * lC[e] + sigf(gi) * tanhf(gg);
      float h  = sigf(go) * tanhf(c);
      lC[e] = c;
      lH[m * 64 + n] = (_Float16)h;
      if (LAYER == 0) hs_out[(size_t)((t * BB + m0 + m) * HH + n)] = (_Float16)h;
      if (t == TT - 1) {
        outH[(size_t)((m0 + m) * HH + n)] = h;
        outC[(size_t)((m0 + m) * HH + n)] = c;
      }
    }
    __syncthreads();

    // ---- fused output projection (layer 1 only): y = h2 @ Wout^T + b -------
    if (LAYER == 1) {
      v16h Ay0 = fragA(lH + row * 64, 0, koff);
      v16h Ay1 = fragA(lH + row * 64, 32, koff);
      v8f a0 = {}, a1 = {};
      a0 = wmma_f16(Ay0, Bo00, a0);
      a0 = wmma_f16(Ay1, Bo01, a0);
      a1 = wmma_f16(Ay0, Bo10, a1);
      a1 = wmma_f16(Ay1, Bo11, a1);
      if (nloc < OUT_) {               // columns 0..15 always valid
        float bo = lbo[nloc];
#pragma unroll
        for (int r = 0; r < 8; ++r)
          y[(size_t)((t * BB + m0 + mb + r) * OUT_ + nloc)] = a0[r] + bo;
      }
      if (16 + nloc < OUT_) {
        float bo = lbo[16 + nloc];
#pragma unroll
        for (int r = 0; r < 8; ++r)
          y[(size_t)((t * BB + m0 + mb + r) * OUT_ + 16 + nloc)] = a1[r] + bo;
      }
      // lH reads above are ordered against next step's writes by the barriers
      // inside the next iteration (gate-store __syncthreads).
    }
  }
}

// ---------------------------------------------------------------------------
// Workspace layout (bytes):
//   0       wc0   160*96  f16  (30720)
//   30720   wc1   160*128 f16  (40960)
//   71680   wo    32*64   f16  (4096)
//   75776   b0    160 f32      (640)
//   76416   b1    160 f32      (640)
//   77056   bo    32  f32      (128)
//   77312   hs1   T*B*H  f16   (83,886,080)
// ---------------------------------------------------------------------------
extern "C" void kernel_launch(void* const* d_in, const int* in_sizes, int n_in,
                              void* d_out, int out_size, void* d_ws, size_t ws_size,
                              hipStream_t stream) {
  const float* x    = (const float*)d_in[0];
  const float* Wih0 = (const float*)d_in[1];
  const float* Whh0 = (const float*)d_in[2];
  const float* bih0 = (const float*)d_in[3];
  const float* bhh0 = (const float*)d_in[4];
  const float* Wih1 = (const float*)d_in[5];
  const float* Whh1 = (const float*)d_in[6];
  const float* bih1 = (const float*)d_in[7];
  const float* bhh1 = (const float*)d_in[8];
  const float* Wout = (const float*)d_in[9];
  const float* bout = (const float*)d_in[10];

  char* ws = (char*)d_ws;
  _Float16* wc0 = (_Float16*)(ws + 0);
  _Float16* wc1 = (_Float16*)(ws + 30720);
  _Float16* wo  = (_Float16*)(ws + 71680);
  float*    b0  = (float*)(ws + 75776);
  float*    b1  = (float*)(ws + 76416);
  float*    bo  = (float*)(ws + 77056);
  _Float16* hs1 = (_Float16*)(ws + 77312);

  float* y    = (float*)d_out;
  float* outH = y + (size_t)TT * BB * OUT_;           // (2,B,H)
  float* outC = outH + 2 * (size_t)BB * HH;           // (2,B,H)

  prep_kernel<<<40, 256, 0, stream>>>(Wih0, Whh0, bih0, bhh0,
                                      Wih1, Whh1, bih1, bhh1,
                                      Wout, bout, wc0, wc1, wo, b0, b1, bo);

  lstm_layer_kernel<0><<<BB / 16, 32, 0, stream>>>(
      x, nullptr, wc0, b0, nullptr, nullptr, hs1, nullptr,
      outH + 0 * (size_t)BB * HH, outC + 0 * (size_t)BB * HH);

  lstm_layer_kernel<1><<<BB / 16, 32, 0, stream>>>(
      nullptr, hs1, wc1, b1, wo, bo, nullptr, y,
      outH + 1 * (size_t)BB * HH, outC + 1 * (size_t)BB * HH);
}
